// PointNet2Classification_34239479284302
// MI455X (gfx1250) — matrix operand order repaired and checked by reading, not compiled, not run on hardware
//
#include <hip/hip_runtime.h>
#include <hip/hip_bf16.h>

typedef __attribute__((ext_vector_type(16))) _Float16     v16h;
typedef __attribute__((ext_vector_type(8)))  _Float16     h8;
typedef __attribute__((ext_vector_type(8)))  float        v8f;
typedef __attribute__((ext_vector_type(4)))  unsigned int v4u;

#define BATCH   8
#define NPTS0   2048
#define M1PTS   1024
#define M2PTS   256
#define KNBR    64
#define R1EDGES (BATCH * M1PTS * KNBR)   // 524288
#define R2EDGES (BATCH * M2PTS * KNBR)   // 131072
#define R3ROWS  (BATCH * M2PTS)          // 2048

static inline int ceil_div_i(int a, int b) { return (a + b - 1) / b; }

// ---------------------------------------------------------------- utilities
__global__ void zero_f32(float* __restrict__ p, int n) {
  int i = blockIdx.x * blockDim.x + threadIdx.x;
  if (i < n) p[i] = 0.f;
}

__global__ void cvt_f32_to_f16(const float* __restrict__ in, _Float16* __restrict__ out, int n) {
  int i = blockIdx.x * blockDim.x + threadIdx.x;
  if (i < n) out[i] = (_Float16)in[i];
}

// Repack W[Cin,Cout] (f32, row-major) into WMMA B-fragment order:
// dword[((ct*nkt + kt)*32 + lane)*8 + v] = pack_f16(W[kb,col], W[kb+1,col])
// with col = ct*16 + (lane&15), kb = kt*32 + 8*(lane>>4) + 2*(v&3) + 16*(v>>2).
// Out-of-range cols/rows packed as zero -> GEMM needs no K/N guards.
__global__ void repack_w(const float* __restrict__ Wf, int Cin, int Cout,
                         int nkt, int total, unsigned int* __restrict__ Wpk) {
  int gid = blockIdx.x * blockDim.x + threadIdx.x;
  if (gid >= total) return;
  const int v    = gid & 7;
  const int lane = (gid >> 3) & 31;
  const int ktI  = (gid >> 8) % nkt;
  const int ct   = (gid >> 8) / nkt;
  const int col  = ct * 16 + (lane & 15);
  const int kb   = ktI * 32 + 8 * (lane >> 4) + 2 * (v & 3) + 16 * (v >> 2);
  unsigned int lo = 0, hi = 0;
  if (col < Cout) {
    if (kb     < Cin) lo = (unsigned int)__builtin_bit_cast(unsigned short, (_Float16)Wf[(size_t)kb * Cout + col]);
    if (kb + 1 < Cin) hi = (unsigned int)__builtin_bit_cast(unsigned short, (_Float16)Wf[(size_t)(kb + 1) * Cout + col]);
  }
  Wpk[gid] = lo | (hi << 16);
}

// ---------------------------------------------------------------- FPS
__global__ void fps_kernel(const float* __restrict__ pos, int Npts, int M,
                           int* __restrict__ idxOut, float* __restrict__ centers) {
  __shared__ float mind[2048];
  __shared__ float rv[256];
  __shared__ int   ri[256];
  const int b = blockIdx.x;
  const int t = threadIdx.x;
  const float* P = pos + (size_t)b * Npts * 3;
  for (int i = t; i < Npts; i += 256) mind[i] = 3.4e38f;
  __syncthreads();
  int last = 0;
  for (int m = 0; m < M; ++m) {
    const float cx = P[last * 3 + 0], cy = P[last * 3 + 1], cz = P[last * 3 + 2];
    if (t == 0) {
      idxOut[b * M + m] = last;
      centers[((size_t)b * M + m) * 3 + 0] = cx;
      centers[((size_t)b * M + m) * 3 + 1] = cy;
      centers[((size_t)b * M + m) * 3 + 2] = cz;
    }
    float bv = -1.0f; int bi = 0;
    for (int i = t; i < Npts; i += 256) {
      float dx = P[i * 3 + 0] - cx, dy = P[i * 3 + 1] - cy, dz = P[i * 3 + 2] - cz;
      float d = dx * dx + dy * dy + dz * dz;
      float md = fminf(mind[i], d);
      mind[i] = md;
      if (md > bv) { bv = md; bi = i; }
    }
    rv[t] = bv; ri[t] = bi;
    __syncthreads();
    for (int s = 128; s > 0; s >>= 1) {
      if (t < s) {
        if (rv[t + s] > rv[t] || (rv[t + s] == rv[t] && ri[t + s] < ri[t])) {
          rv[t] = rv[t + s]; ri[t] = ri[t + s];
        }
      }
      __syncthreads();
    }
    last = ri[0];
    __syncthreads();
  }
}

// ---------------------------------------------------------------- ball query
__global__ void ball_query(const float* __restrict__ pos, const float* __restrict__ centers,
                           int Npts, int M, int nCenters, float r2,
                           int* __restrict__ nbr, float* __restrict__ mask,
                           float* __restrict__ cntAccum) {
  const int wid  = (blockIdx.x * blockDim.x + threadIdx.x) >> 5;
  const int lane = threadIdx.x & 31;
  if (wid >= nCenters) return;
  const int b = wid / M;
  const float* P = pos + (size_t)b * Npts * 3;
  const float cx = centers[(size_t)wid * 3 + 0];
  const float cy = centers[(size_t)wid * 3 + 1];
  const float cz = centers[(size_t)wid * 3 + 2];
  int*   nb = nbr  + (size_t)wid * KNBR;
  float* mk = mask + (size_t)wid * KNBR;
  int cnt = 0;
  for (int base = 0; base < Npts && cnt < KNBR; base += 32) {
    const int j = base + lane;
    bool in = false;
    if (j < Npts) {
      float dx = P[j * 3 + 0] - cx, dy = P[j * 3 + 1] - cy, dz = P[j * 3 + 2] - cz;
      in = (dx * dx + dy * dy + dz * dz) <= r2;
    }
    unsigned bal = (unsigned)__ballot(in);
    int posn = cnt + __popc(bal & ((lane == 0) ? 0u : ((1u << lane) - 1u)));
    if (in && posn < KNBR) { nb[posn] = j; mk[posn] = 1.f; }
    cnt += __popc(bal);
  }
  int valid = cnt < KNBR ? cnt : KNBR;
  for (int k = valid + lane; k < KNBR; k += 32) { nb[k] = 0; mk[k] = 0.f; }
  if (lane == 0) atomicAdd(cntAccum, (float)valid);
}

// ---------------------------------------------------------------- features (K-padded)
__global__ void feat_sa1(const float* __restrict__ x, const float* __restrict__ pos,
                         const float* __restrict__ centers, const int* __restrict__ nbr,
                         int total, _Float16* __restrict__ F) {   // lda = 32
  int e = blockIdx.x * blockDim.x + threadIdx.x;
  if (e >= total) return;
  const int cen = e / KNBR;
  const int b = cen / M1PTS;
  const int j = nbr[e];
  const float* X = x   + ((size_t)b * NPTS0 + j) * 3;
  const float* P = pos + ((size_t)b * NPTS0 + j) * 3;
  const float* C = centers + (size_t)cen * 3;
  _Float16* f = F + (size_t)e * 32;
  f[0] = (_Float16)X[0]; f[1] = (_Float16)X[1]; f[2] = (_Float16)X[2];
  f[3] = (_Float16)(P[0] - C[0]); f[4] = (_Float16)(P[1] - C[1]); f[5] = (_Float16)(P[2] - C[2]);
#pragma unroll
  for (int c = 6; c < 32; ++c) f[c] = (_Float16)0.f;
}

__global__ void feat_gather(const float* __restrict__ feat, const float* __restrict__ posPrev,
                            const float* __restrict__ centers, const int* __restrict__ nbr,
                            int Mprev, int M, int Cf, int Ct, int total, _Float16* __restrict__ F) {
  int gid = blockIdx.x * blockDim.x + threadIdx.x;
  if (gid >= total) return;
  const int e = gid / Ct;
  const int c = gid % Ct;
  const int cen = e / KNBR;
  const int b = cen / M;
  const int j = nbr[e];
  float v = 0.f;
  if (c < Cf)            v = feat[((size_t)b * Mprev + j) * Cf + c];
  else if (c < Cf + 3)   v = posPrev[((size_t)b * Mprev + j) * 3 + (c - Cf)]
                           - centers[(size_t)cen * 3 + (c - Cf)];
  F[(size_t)e * Ct + c] = (_Float16)v;
}

__global__ void concat_h_pos(const float* __restrict__ h2, const float* __restrict__ pos2,
                             int total, _Float16* __restrict__ F) {   // lda = 288
  int gid = blockIdx.x * blockDim.x + threadIdx.x;
  if (gid >= total) return;
  const int e = gid / 288;
  const int c = gid % 288;
  float v = 0.f;
  if (c < 256)      v = h2[(size_t)e * 256 + c];
  else if (c < 259) v = pos2[(size_t)e * 3 + (c - 256)];
  F[gid] = (_Float16)v;
}

// ---------------------------------------------------------------- WMMA GEMM
// Y[R,Cout] = transform(A[R,Cinp]) @ W + bias;  Cinp % 32 == 0 (zero-padded).
// MODE: 0 = raw, 1 = ReLU, 2 = BN(f16 scale/shift via LDS, packed pk_fma)+ReLU.
// Software-pipelined: tile ki+1 fragments are loaded before the wmma on tile ki,
// so the wmma only waits on older loads and load latency overlaps the matrix op.
template <int MODE>
__global__ void __launch_bounds__(256)
wmma_gemm(const _Float16* __restrict__ A, int R, int Cinp, int Cout,
          const unsigned int* __restrict__ Wpk, const float* __restrict__ bias,
          const _Float16* __restrict__ bnScaleH, const _Float16* __restrict__ bnShiftH,
          const float* __restrict__ rowMask,
          float* __restrict__ statSum, float* __restrict__ statSsq,
          _Float16* __restrict__ Yh, float* __restrict__ Yf) {
  __shared__ __align__(16) _Float16 sScale[1024];
  __shared__ __align__(16) _Float16 sShift[1024];
  __shared__ float sSum[16];
  __shared__ float sSsq[16];

  const int tid    = threadIdx.x;
  const int wave   = tid >> 5;
  const int lane   = tid & 31;
  const int halfId = lane >> 4;
  const int l16    = lane & 15;

  if (MODE == 2) {
    for (int i = tid; i < Cinp; i += 256) { sScale[i] = bnScaleH[i]; sShift[i] = bnShiftH[i]; }
  }
  if (tid < 16) { sSum[tid] = 0.f; sSsq[tid] = 0.f; }
  __syncthreads();

  const int rowTile0 = blockIdx.x * 128 + wave * 16;
  const int colTile0 = blockIdx.y * 16;
  const int aRow   = rowTile0 + l16;
  const int bCol   = colTile0 + l16;
  const int aRowC  = aRow < R ? aRow : (R - 1);    // clamp; OOR rows never stored
  const int nkt    = Cinp >> 5;

  const v4u* __restrict__ Ar = (const v4u*)(A + (size_t)aRowC * Cinp);
  const v4u* __restrict__ Bt = (const v4u*)Wpk + (((size_t)blockIdx.y * nkt) * 32 + lane) * 2;

  // prologue: load tile 0
  v4u ca0 = Ar[halfId];
  v4u ca1 = Ar[2 + halfId];
  v4u cb0 = Bt[0];
  v4u cb1 = Bt[1];

  v8f acc = {};
  for (int ki = 0; ki < nkt; ++ki) {
    const int kn = (ki + 1 < nkt) ? (ki + 1) : ki;   // clamped re-load on last iter
    if (ki + 2 < nkt) __builtin_prefetch(&Ar[(ki + 2) * 4], 0, 1);
    // issue next-tile loads before consuming current tile
    v4u na0 = Ar[kn * 4 + halfId];
    v4u na1 = Ar[kn * 4 + 2 + halfId];
    const v4u* Bn = Bt + (size_t)kn * 64;
    v4u nb0 = Bn[0];
    v4u nb1 = Bn[1];

    h8 al = __builtin_bit_cast(h8, ca0);
    h8 ah = __builtin_bit_cast(h8, ca1);
    if (MODE == 2) {
      const int k0 = ki * 32 + 8 * halfId;   // al covers k0..k0+7 (16B-aligned)
      const int k1 = k0 + 16;                // ah covers k1..k1+7
      const h8 sc0 = *(const h8*)&sScale[k0];
      const h8 sf0 = *(const h8*)&sShift[k0];
      const h8 sc1 = *(const h8*)&sScale[k1];
      const h8 sf1 = *(const h8*)&sShift[k1];
      al = al * sc0 + sf0;                   // v_pk_fma_f16
      ah = ah * sc1 + sf1;
    }
    if (MODE >= 1) {
      const h8 z = (h8)(_Float16)0.f;
      al = __builtin_elementwise_max(al, z); // v_pk_max
      ah = __builtin_elementwise_max(ah, z);
    }
    v16h av = __builtin_shufflevector(al, ah, 0, 1, 2, 3, 4, 5, 6, 7,
                                              8, 9, 10, 11, 12, 13, 14, 15);
    v16h bv = __builtin_shufflevector(__builtin_bit_cast(h8, cb0), __builtin_bit_cast(h8, cb1),
                                      0, 1, 2, 3, 4, 5, 6, 7, 8, 9, 10, 11, 12, 13, 14, 15);
    acc = __builtin_amdgcn_wmma_f32_16x16x32_f16(false, av, false, bv, (short)0, acc, false, false);
    ca0 = na0; ca1 = na1; cb0 = nb0; cb1 = nb1;
  }

  const bool bColOk = bCol < Cout;
  const float bia = bColOk ? bias[bCol] : 0.f;
  float lsum = 0.f, lssq = 0.f;
#pragma unroll
  for (int r = 0; r < 8; ++r) {
    const int row = rowTile0 + r + 8 * halfId;   // C/D: vgpr r -> M=r (+8 hi half)
    if (row < R && bColOk) {
      const float val = acc[r] + bia;
      if (Yh) Yh[(size_t)row * Cout + bCol] = (_Float16)val;
      if (Yf) Yf[(size_t)row * Cout + bCol] = val;
      if (statSum) {
        const float w = rowMask ? rowMask[row] : 1.f;
        lsum += val * w;
        lssq += val * val * w;
      }
    }
  }
  if (statSum) {
    atomicAdd(&sSum[l16], lsum);
    atomicAdd(&sSsq[l16], lssq);
    __syncthreads();
    if (tid < 16 && colTile0 + tid < Cout) {
      atomicAdd(&statSum[colTile0 + tid], sSum[tid]);
      atomicAdd(&statSsq[colTile0 + tid], sSsq[tid]);
    }
  }
}

// ---------------------------------------------------------------- BN finalize
__global__ void bn_finalize(const float* __restrict__ statSum, const float* __restrict__ statSsq,
                            const float* __restrict__ gamma, const float* __restrict__ beta,
                            const float* __restrict__ cntPtr, float cntConst,
                            _Float16* __restrict__ scaleH, _Float16* __restrict__ shiftH, int C) {
  int c = blockIdx.x * blockDim.x + threadIdx.x;
  if (c >= C) return;
  const float cnt  = cntPtr ? cntPtr[0] : cntConst;
  const float mean = statSum[c] / cnt;
  float var = statSsq[c] / cnt - mean * mean;
  var = fmaxf(var, 0.f);
  const float sc = gamma[c] * rsqrtf(var + 1e-5f);
  scaleH[c] = (_Float16)sc;
  shiftH[c] = (_Float16)(beta[c] - mean * sc);
}

// ---------------------------------------------------------------- pooling
__global__ void maxpool_edges(const _Float16* __restrict__ Y, const float* __restrict__ mask,
                              int C, int total, float* __restrict__ out) {
  int gid = blockIdx.x * blockDim.x + threadIdx.x;
  if (gid >= total) return;
  const int c   = gid % C;
  const int cen = gid / C;
  const _Float16* y = Y + (size_t)cen * KNBR * C + c;
  const float* mk = mask + (size_t)cen * KNBR;
  float best = -3.4e38f;
  for (int k = 0; k < KNBR; ++k)
    if (mk[k] != 0.f) best = fmaxf(best, (float)y[(size_t)k * C]);
  out[gid] = best;
}

__global__ void global_maxpool(const _Float16* __restrict__ Y, int Mrows, int C,
                               int total, float* __restrict__ out) {
  int gid = blockIdx.x * blockDim.x + threadIdx.x;
  if (gid >= total) return;
  const int b = gid / C;
  const int c = gid % C;
  const _Float16* y = Y + (size_t)b * Mrows * C + c;
  float best = -3.4e38f;
  for (int m = 0; m < Mrows; ++m) best = fmaxf(best, (float)y[(size_t)m * C]);
  out[gid] = best;
}

// ---------------------------------------------------------------- launcher
extern "C" void kernel_launch(void* const* d_in, const int* in_sizes, int n_in,
                              void* d_out, int out_size, void* d_ws, size_t ws_size,
                              hipStream_t stream) {
  (void)in_sizes; (void)n_in; (void)out_size; (void)ws_size;

  const float* x   = (const float*)d_in[0];
  const float* pos = (const float*)d_in[1];
  // params in reference-file dict insertion order: sa1, sa2, sa3, head; lin then bn.
  const float* Wf[12]; const float* Bf[12];
  const float* G[6];  const float* Be[6];
  int wi = 0, gi = 0, p = 2;
  for (int mod = 0; mod < 4; ++mod) {
    for (int l = 0; l < 3; ++l) { Wf[wi] = (const float*)d_in[p++]; Bf[wi] = (const float*)d_in[p++]; ++wi; }
    if (mod < 3) {
      for (int l = 0; l < 2; ++l) { G[gi] = (const float*)d_in[p++]; Be[gi] = (const float*)d_in[p++]; ++gi; }
    }
  }
  static const int Cin_[12]  = {6, 64, 64, 131, 128, 128, 259, 256, 512, 1024, 512, 256};
  static const int Cinp_[12] = {32, 64, 64, 160, 128, 128, 288, 256, 512, 1024, 512, 256};
  static const int Cout_[12] = {64, 64, 128, 128, 128, 256, 256, 512, 1024, 512, 256, 40};

  // ---- workspace layout
  char* wp = (char*)d_ws;
  auto alloc = [&](size_t bytes) -> char* {
    char* r = wp; wp += (bytes + 255) & ~(size_t)255; return r;
  };
  unsigned int* Wpk[12];
  for (int i = 0; i < 12; ++i) {
    size_t dwords = (size_t)ceil_div_i(Cout_[i], 16) * (Cinp_[i] / 32) * 256;
    Wpk[i] = (unsigned int*)alloc(dwords * 4);
  }
  int*   idx1  = (int*)  alloc((size_t)BATCH * M1PTS * 4);
  float* pos1  = (float*)alloc((size_t)BATCH * M1PTS * 3 * 4);
  int*   idx2  = (int*)  alloc((size_t)BATCH * M2PTS * 4);
  float* pos2  = (float*)alloc((size_t)BATCH * M2PTS * 3 * 4);
  int*   nbr1  = (int*)  alloc((size_t)R1EDGES * 4);
  float* mask1 = (float*)alloc((size_t)R1EDGES * 4);
  int*   nbr2  = (int*)  alloc((size_t)R2EDGES * 4);
  float* mask2 = (float*)alloc((size_t)R2EDGES * 4);
  float* cnt   = (float*)alloc(4 * 4);
  float* stats = (float*)alloc(2048 * 4);
  float* ssum  = stats; float* sssq = stats + 1024;
  _Float16* bnscH = (_Float16*)alloc(1024 * 2);
  _Float16* bnshH = (_Float16*)alloc(1024 * 2);
  _Float16* F1    = (_Float16*)alloc((size_t)R1EDGES * 32 * 2);
  _Float16* ACT_A = (_Float16*)alloc((size_t)R1EDGES * 128 * 2);  // 134 MB (fits L2)
  _Float16* ACT_B = (_Float16*)alloc((size_t)R1EDGES * 64 * 2);   // 67 MB
  float* h1 = (float*)alloc((size_t)BATCH * M1PTS * 128 * 4);
  float* h2 = (float*)alloc((size_t)BATCH * M2PTS * 256 * 4);
  _Float16* F3  = (_Float16*)alloc((size_t)R3ROWS * 288 * 2);
  _Float16* Y3a = (_Float16*)alloc((size_t)R3ROWS * 256 * 2);
  _Float16* Y3b = (_Float16*)alloc((size_t)R3ROWS * 512 * 2);
  _Float16* Y3c = (_Float16*)alloc((size_t)R3ROWS * 1024 * 2);
  float*    gfeat  = (float*)alloc((size_t)BATCH * 1024 * 4);
  _Float16* gfeath = (_Float16*)alloc((size_t)BATCH * 1024 * 2);
  _Float16* Yh0 = (_Float16*)alloc((size_t)BATCH * 512 * 2);
  _Float16* Yh1 = (_Float16*)alloc((size_t)BATCH * 256 * 2);

  auto gemm = [&](const _Float16* A, int R, int l, int mode, const float* mask,
                  bool doStats, _Float16* Yhout, float* Yfout) {
    if (doStats) zero_f32<<<ceil_div_i(2048, 256), 256, 0, stream>>>(stats, 2048);
    dim3 g(ceil_div_i(R, 128), ceil_div_i(Cout_[l], 16));
    float* su = doStats ? ssum : nullptr;
    float* sq = doStats ? sssq : nullptr;
    if (mode == 0)
      wmma_gemm<0><<<g, 256, 0, stream>>>(A, R, Cinp_[l], Cout_[l], Wpk[l], Bf[l],
                                          nullptr, nullptr, mask, su, sq, Yhout, Yfout);
    else if (mode == 1)
      wmma_gemm<1><<<g, 256, 0, stream>>>(A, R, Cinp_[l], Cout_[l], Wpk[l], Bf[l],
                                          nullptr, nullptr, mask, su, sq, Yhout, Yfout);
    else
      wmma_gemm<2><<<g, 256, 0, stream>>>(A, R, Cinp_[l], Cout_[l], Wpk[l], Bf[l],
                                          bnscH, bnshH, mask, su, sq, Yhout, Yfout);
  };
  auto finalize = [&](int bnIdx, int C, const float* cntPtr, float cntConst) {
    bn_finalize<<<ceil_div_i(C, 256), 256, 0, stream>>>(ssum, sssq, G[bnIdx], Be[bnIdx],
                                                        cntPtr, cntConst, bnscH, bnshH, C);
  };

  // ---- weight repack + init
  for (int i = 0; i < 12; ++i) {
    int nkt = Cinp_[i] / 32;
    int total = ceil_div_i(Cout_[i], 16) * nkt * 256;
    repack_w<<<ceil_div_i(total, 256), 256, 0, stream>>>(Wf[i], Cin_[i], Cout_[i], nkt, total, Wpk[i]);
  }
  zero_f32<<<1, 4, 0, stream>>>(cnt, 4);

  // ================= SA1 =================
  fps_kernel<<<BATCH, 256, 0, stream>>>(pos, NPTS0, M1PTS, idx1, pos1);
  ball_query<<<ceil_div_i(BATCH * M1PTS * 32, 256), 256, 0, stream>>>(
      pos, pos1, NPTS0, M1PTS, BATCH * M1PTS, 0.2f * 0.2f, nbr1, mask1, cnt + 0);
  feat_sa1<<<ceil_div_i(R1EDGES, 256), 256, 0, stream>>>(x, pos, pos1, nbr1, R1EDGES, F1);

  gemm(F1, R1EDGES, 0, 0, mask1, true, ACT_A, nullptr);
  finalize(0, 64, cnt + 0, 0.f);
  gemm(ACT_A, R1EDGES, 1, 2, mask1, true, ACT_B, nullptr);
  finalize(1, 64, cnt + 0, 0.f);
  gemm(ACT_B, R1EDGES, 2, 2, nullptr, false, ACT_A, nullptr);
  maxpool_edges<<<ceil_div_i(BATCH * M1PTS * 128, 256), 256, 0, stream>>>(
      ACT_A, mask1, 128, BATCH * M1PTS * 128, h1);

  // ================= SA2 =================
  fps_kernel<<<BATCH, 256, 0, stream>>>(pos1, M1PTS, M2PTS, idx2, pos2);
  ball_query<<<ceil_div_i(BATCH * M2PTS * 32, 256), 256, 0, stream>>>(
      pos1, pos2, M1PTS, M2PTS, BATCH * M2PTS, 0.4f * 0.4f, nbr2, mask2, cnt + 1);
  _Float16* F2 = ACT_B;  // reuse (R2*160 f16 < R1*64 f16)
  {
    int total = R2EDGES * 160;
    feat_gather<<<ceil_div_i(total, 256), 256, 0, stream>>>(h1, pos1, pos2, nbr2,
                                                            M1PTS, M2PTS, 128, 160, total, F2);
  }
  gemm(F2, R2EDGES, 3, 0, mask2, true, ACT_A, nullptr);
  finalize(2, 128, cnt + 1, 0.f);
  gemm(ACT_A, R2EDGES, 4, 2, mask2, true, ACT_B, nullptr);
  finalize(3, 128, cnt + 1, 0.f);
  gemm(ACT_B, R2EDGES, 5, 2, nullptr, false, ACT_A, nullptr);
  maxpool_edges<<<ceil_div_i(BATCH * M2PTS * 256, 256), 256, 0, stream>>>(
      ACT_A, mask2, 256, BATCH * M2PTS * 256, h2);

  // ================= SA3 (global MLP + max pool) =================
  concat_h_pos<<<ceil_div_i(R3ROWS * 288, 256), 256, 0, stream>>>(h2, pos2, R3ROWS * 288, F3);
  gemm(F3, R3ROWS, 6, 0, nullptr, true, Y3a, nullptr);
  finalize(4, 256, nullptr, (float)R3ROWS);
  gemm(Y3a, R3ROWS, 7, 2, nullptr, true, Y3b, nullptr);
  finalize(5, 512, nullptr, (float)R3ROWS);
  gemm(Y3b, R3ROWS, 8, 2, nullptr, false, Y3c, nullptr);
  global_maxpool<<<ceil_div_i(BATCH * 1024, 256), 256, 0, stream>>>(Y3c, M2PTS, 1024,
                                                                    BATCH * 1024, gfeat);
  cvt_f32_to_f16<<<ceil_div_i(BATCH * 1024, 256), 256, 0, stream>>>(gfeat, gfeath, BATCH * 1024);

  // ================= Head =================
  gemm(gfeath, BATCH, 9,  0, nullptr, false, Yh0, nullptr);
  gemm(Yh0,    BATCH, 10, 1, nullptr, false, Yh1, nullptr);
  gemm(Yh1,    BATCH, 11, 1, nullptr, false, nullptr, (float*)d_out);
}
